// JRTransformer_33509334843568
// MI455X (gfx1250) — matrix operand order repaired and checked by reading, not compiled, not run on hardware
//
#include <hip/hip_runtime.h>
#include <hip/hip_bf16.h>

#define BB 4
#define NN 128
#define CC 128
#define HH 8
#define HSZ 16
#define DD 2
#define ATT_SCALE 0.25f   // HS^-0.5

typedef __bf16 bf16_t;
typedef __attribute__((ext_vector_type(16))) __bf16 v16bf;
typedef __attribute__((ext_vector_type(8)))  __bf16 v8bf;
typedef __attribute__((ext_vector_type(8)))  float  v8f;

__device__ __forceinline__ float geluf(float x) {
  return 0.5f * x * (1.0f + erff(x * 0.7071067811865476f));
}

// ---------------- LayerNorm (wave-per-row, wave32 shuffles) ----------------
template<int CW>
__global__ void ln_rows(const float* __restrict__ in, const float* __restrict__ g,
                        const float* __restrict__ be, bf16_t* __restrict__ out, int rows) {
  int wid  = threadIdx.x >> 5;
  int lane = threadIdx.x & 31;
  int row  = blockIdx.x * (blockDim.x >> 5) + wid;
  if (row >= rows) return;
  const float* p = in + (size_t)row * CW;
  float v[CW / 32];
  float s = 0.f, sq = 0.f;
#pragma unroll
  for (int t = 0; t < CW / 32; ++t) {
    float x = p[lane + 32 * t];
    v[t] = x; s += x; sq += x * x;
  }
#pragma unroll
  for (int off = 16; off; off >>= 1) {
    s  += __shfl_xor(s,  off, 32);
    sq += __shfl_xor(sq, off, 32);
  }
  float mean = s * (1.0f / CW);
  float var  = sq * (1.0f / CW) - mean * mean;
  float inv  = rsqrtf(var + 1e-5f);
  bf16_t* q = out + (size_t)row * CW;
#pragma unroll
  for (int t = 0; t < CW / 32; ++t) {
    int c = lane + 32 * t;
    q[c] = (bf16_t)((v[t] - mean) * inv * g[c] + be[c]);
  }
}

// -------- build concat(r_ij, r_ji, jf_j, jf_i) and LayerNorm -> bf16 -------
__global__ void build_rin_ln(const float* __restrict__ rf, const float* __restrict__ jf,
                             const float* __restrict__ g, const float* __restrict__ be,
                             bf16_t* __restrict__ out) {
  int wid  = threadIdx.x >> 5;
  int lane = threadIdx.x & 31;
  int row  = blockIdx.x * 8 + wid;            // row = (b*128+i)*128+j
  int b = row >> 14, i = (row >> 7) & 127, j = row & 127;
  const float* r_ij = rf + (size_t)row * CC;
  const float* r_ji = rf + ((size_t)((b * NN + j) * NN + i)) * CC;
  const float* j_j  = jf + (size_t)(b * NN + j) * CC;
  const float* j_i  = jf + (size_t)(b * NN + i) * CC;
  float v[16];
  float s = 0.f, sq = 0.f;
#pragma unroll
  for (int t = 0; t < 16; ++t) {
    int c = lane + 32 * t;
    float x;
    if      (c < 128) x = r_ij[c];
    else if (c < 256) x = r_ji[c - 128];
    else if (c < 384) x = j_j[c - 256];
    else              x = j_i[c - 384];
    v[t] = x; s += x; sq += x * x;
  }
#pragma unroll
  for (int off = 16; off; off >>= 1) {
    s  += __shfl_xor(s,  off, 32);
    sq += __shfl_xor(sq, off, 32);
  }
  float mean = s * (1.0f / 512.0f);
  float var  = sq * (1.0f / 512.0f) - mean * mean;
  float inv  = rsqrtf(var + 1e-5f);
  bf16_t* q = out + (size_t)row * 512;
#pragma unroll
  for (int t = 0; t < 16; ++t) {
    int c = lane + 32 * t;
    q[c] = (bf16_t)((v[t] - mean) * inv * g[c] + be[c]);
  }
}

// -------- pack f32 [K,N] weight into WMMA B-fragment bf16 layout -----------
// idx = ((kt*ntiles + nt)*32 + lane)*16 + e ; k = kt*32 + (lane>>4)*8 + (e&7) + (e>>3)*16 ;
// n = nt*16 + (lane&15). Zero-pad n >= N (ntiles is padded to an even count).
__global__ void pack_w(const float* __restrict__ W, bf16_t* __restrict__ P,
                       int K, int N, int ntiles) {
  int idx = blockIdx.x * blockDim.x + threadIdx.x;
  int total = (K / 32) * ntiles * 512;
  if (idx >= total) return;
  int e    = idx & 15;
  int lane = (idx >> 4) & 31;
  int rest = idx >> 9;
  int nt   = rest % ntiles;
  int kt   = rest / ntiles;
  int k = kt * 32 + ((lane >> 4) << 3) + (e & 7) + ((e >> 3) << 4);
  int n = nt * 16 + (lane & 15);
  float val = (n < N) ? W[(size_t)k * N + n] : 0.0f;
  P[idx] = (bf16_t)val;
}

// ------------------------- WMMA bf16 GEMM (32x32 per wave) -----------------
// One wave = 2x2 grid of 16x16 D-tiles; K in steps of 32. Per K-step:
// 2 A fragments + 2 B fragments -> 4 x v_wmma_f32_16x16x32_bf16 (2x operand
// reuse in registers, ~16 FLOP/byte from L2).
template<bool GELU, bool RES, bool OUTBF>
__global__ void gemm_wmma(const bf16_t* __restrict__ A, const bf16_t* __restrict__ Bp,
                          const float* __restrict__ bias, const float* __restrict__ res,
                          float* __restrict__ outF, bf16_t* __restrict__ outB,
                          int M, int N, int K, int ntiles /* even */) {
  int lane   = threadIdx.x;                    // 0..31
  int npair  = blockIdx.x * blockDim.y + threadIdx.y;
  int npairs = ntiles >> 1;
  if (npair >= npairs) return;                 // wave-uniform exit (EXEC all-1 for WMMA)
  int mbase = blockIdx.y * 32;
  int hi = lane >> 4, lo = lane & 15;

  const bf16_t* arow0 = A + (size_t)(mbase + lo) * K + hi * 8;
  const bf16_t* arow1 = A + (size_t)(mbase + 16 + lo) * K + hi * 8;
  const bf16_t* bptr  = Bp + ((size_t)(npair * 2) * 32 + lane) * 16;
  size_t bstride = (size_t)ntiles * 512;

  v8f acc00 = {}, acc01 = {}, acc10 = {}, acc11 = {};
  for (int k0 = 0; k0 < K; k0 += 32) {
    v8bf a0lo = *(const v8bf*)(arow0 + k0);
    v8bf a0hi = *(const v8bf*)(arow0 + k0 + 16);
    v8bf a1lo = *(const v8bf*)(arow1 + k0);
    v8bf a1hi = *(const v8bf*)(arow1 + k0 + 16);
    v16bf a0, a1;
#pragma unroll
    for (int t = 0; t < 8; ++t) {
      a0[t] = a0lo[t]; a0[t + 8] = a0hi[t];
      a1[t] = a1lo[t]; a1[t + 8] = a1hi[t];
    }
    v16bf b0 = *(const v16bf*)bptr;
    v16bf b1 = *(const v16bf*)(bptr + 512);
    bptr += bstride;
    acc00 = __builtin_amdgcn_wmma_f32_16x16x32_bf16(false, a0, false, b0, (short)0, acc00, false, false);
    acc01 = __builtin_amdgcn_wmma_f32_16x16x32_bf16(false, a0, false, b1, (short)0, acc01, false, false);
    acc10 = __builtin_amdgcn_wmma_f32_16x16x32_bf16(false, a1, false, b0, (short)0, acc10, false, false);
    acc11 = __builtin_amdgcn_wmma_f32_16x16x32_bf16(false, a1, false, b1, (short)0, acc11, false, false);
  }

  auto emit = [&](const v8f& acc, int gm0, int gn) {
    bool nok = gn < N;
    float bv = (bias != nullptr && nok) ? bias[gn] : 0.0f;
#pragma unroll
    for (int r = 0; r < 8; ++r) {
      int gm = gm0 + r + hi * 8;               // C/D layout: VGPR r -> M = r + 8*hi
      float val = acc[r] + bv;
      if (GELU) val = geluf(val);
      if (nok) {
        size_t oidx = (size_t)gm * N + gn;
        if (OUTBF) outB[oidx] = (bf16_t)val;
        else       outF[oidx] = (RES ? res[oidx] : 0.0f) + val;
      }
    }
  };
  int gn0 = npair * 32 + lo;
  emit(acc00, mbase,      gn0);
  emit(acc01, mbase,      gn0 + 16);
  emit(acc10, mbase + 16, gn0);
  emit(acc11, mbase + 16, gn0 + 16);
}

// ---------------- attention scores (qk + rconv + RqRk) + softmax -----------
__global__ void attn_softmax(const float* __restrict__ jqkv, const float* __restrict__ rlin,
                             const bf16_t* __restrict__ rqk, float* __restrict__ attn) {
  int idx = blockIdx.x;                  // (b*8+h)*128 + i
  int i = idx & 127, h = (idx >> 7) & 7, b = idx >> 10;
  int j = threadIdx.x;                   // 128 threads
  __shared__ float qs[HSZ];
  __shared__ float red[NN];
  if (j < HSZ) qs[j] = jqkv[(size_t)(b * NN + i) * 384 + h * HSZ + j];
  __syncthreads();
  const float* kp = jqkv + (size_t)(b * NN + j) * 384 + 128 + h * HSZ;
  float qk = 0.f;
#pragma unroll
  for (int d = 0; d < HSZ; ++d) qk += qs[d] * kp[d];
  size_t row = (size_t)(b * NN + i) * NN + j;
  float rl = rlin[row * HH + h];
  const bf16_t* rq = rqk + row * 256 + h * HSZ;
  const bf16_t* rk = rq + 128;
  float qu = 0.f;
#pragma unroll
  for (int d = 0; d < HSZ; ++d) qu += (float)rq[d] * (float)rk[d];
  float s = (qk + rl + qu) * ATT_SCALE;

  red[j] = s; __syncthreads();
  for (int off = 64; off > 0; off >>= 1) {
    if (j < off) red[j] = fmaxf(red[j], red[j + off]);
    __syncthreads();
  }
  float mx = red[0]; __syncthreads();
  float ev = expf(s - mx);
  red[j] = ev; __syncthreads();
  for (int off = 64; off > 0; off >>= 1) {
    if (j < off) red[j] += red[j + off];
    __syncthreads();
  }
  float inv = 1.0f / red[0];
  attn[((size_t)idx) * NN + j] = ev * inv;
}

// --------------------------- o = a @ v (bf16 out) --------------------------
__global__ void attn_out(const float* __restrict__ attn, const float* __restrict__ jqkv,
                         bf16_t* __restrict__ o) {
  int t = blockIdx.x * blockDim.x + threadIdx.x;   // B*N*C
  if (t >= BB * NN * CC) return;
  int c = t & 127, i = (t >> 7) & 127, b = t >> 14;
  int h = c >> 4;
  const float* ap = attn + ((size_t)((b * HH + h) * NN + i)) * NN;
  float acc = 0.f;
  for (int j = 0; j < NN; ++j)
    acc += ap[j] * jqkv[(size_t)(b * NN + j) * 384 + 256 + c];
  o[t] = (bf16_t)acc;
}

// ===========================================================================
extern "C" void kernel_launch(void* const* d_in, const int* in_sizes, int n_in,
                              void* d_out, int out_size, void* d_ws, size_t ws_size,
                              hipStream_t stream) {
  (void)in_sizes; (void)n_in; (void)out_size; (void)ws_size;
  float* jf = (float*)d_out;
  float* rf = jf + (size_t)BB * NN * CC;
  hipMemcpyAsync(jf, d_in[0], sizeof(float) * (size_t)BB * NN * CC,
                 hipMemcpyDeviceToDevice, stream);
  hipMemcpyAsync(rf, d_in[1], sizeof(float) * (size_t)BB * NN * NN * CC,
                 hipMemcpyDeviceToDevice, stream);

  char* ws = (char*)d_ws;
  // packed per-layer weights (bf16 fragment layout, even n-tile counts)
  bf16_t* pw_jqkv  = (bf16_t*)(ws + 0);        //  98304 B (K=128,N=384,nt=24)
  bf16_t* pw_rqk   = (bf16_t*)(ws + 98304);    //  65536 B (K=128,N=256,nt=16)
  bf16_t* pw_rconv = (bf16_t*)(ws + 163840);   //   8192 B (K=128,N=8, nt=2)
  bf16_t* pw_proj  = (bf16_t*)(ws + 172032);   //  32768 B
  bf16_t* pw_j1    = (bf16_t*)(ws + 204800);   //  32768 B
  bf16_t* pw_j2    = (bf16_t*)(ws + 237568);   //  32768 B
  bf16_t* pw_r1a   = (bf16_t*)(ws + 270336);   // 524288 B (K=512,N=512,nt=32)
  bf16_t* pw_r1b   = (bf16_t*)(ws + 794624);   // 131072 B (K=512,N=128,nt=8)
  bf16_t* pw_r2a   = (bf16_t*)(ws + 925696);   //  32768 B
  bf16_t* pw_r2b   = (bf16_t*)(ws + 958464);   //  32768 B
  // small joint-side buffers
  bf16_t* x_ln = (bf16_t*)(ws + 1048576);
  bf16_t* y_bf = (bf16_t*)(ws + 1179648);
  bf16_t* h_bf = (bf16_t*)(ws + 1310720);
  bf16_t* o_bf = (bf16_t*)(ws + 1441792);
  float*  jqkv = (float*) (ws + 1572864);   // [512,384]
  float*  rlin = (float*) (ws + 2359296);   // [65536,8]
  float*  attn = (float*) (ws + 4456448);   // [B,H,N,N]
  // big reusable regions (phase-overlapped)
  bf16_t* rA = (bf16_t*)(ws + 8388608);     // r_ln -> rin_ln -> rn2 (64 MB)
  bf16_t* rB = (bf16_t*)(ws + 75497472);    // rqk  -> h1     -> h2  (64 MB)

  const int RROWS = BB * NN * NN;           // 65536

  auto PACK = [&](const float* W, bf16_t* P, int K, int Nn) {
    int nt  = (Nn + 15) / 16;
    int ntp = (nt + 1) & ~1;                // pad to even tile count
    int total = (K / 32) * ntp * 512;
    pack_w<<<(total + 255) / 256, 256, 0, stream>>>(W, P, K, Nn, ntp);
  };
  auto GEMM = [&](int mode, const bf16_t* A, const bf16_t* Bp, const float* bias,
                  const float* res, float* oF, bf16_t* oB, int M, int Nn, int K) {
    int nt  = (Nn + 15) / 16;
    int ntp = (nt + 1) & ~1;
    int npairs = ntp >> 1;
    dim3 blk(32, 4), grd((npairs + 3) / 4, M / 32);
    switch (mode) {
      case 0: gemm_wmma<false, false, false><<<grd, blk, 0, stream>>>(A, Bp, bias, res, oF, oB, M, Nn, K, ntp); break;
      case 1: gemm_wmma<false, false, true ><<<grd, blk, 0, stream>>>(A, Bp, bias, res, oF, oB, M, Nn, K, ntp); break;
      case 2: gemm_wmma<true,  false, true ><<<grd, blk, 0, stream>>>(A, Bp, bias, res, oF, oB, M, Nn, K, ntp); break;
      default:gemm_wmma<false, true,  false><<<grd, blk, 0, stream>>>(A, Bp, bias, res, oF, oB, M, Nn, K, ntp); break;
    }
  };

  for (int dL = 0; dL < DD; ++dL) {
    const float* W_jqkv  = (const float*)d_in[2]  + (size_t)dL * CC * 3 * CC;
    const float* W_rconv = (const float*)d_in[3]  + (size_t)dL * CC * HH;
    const float* W_rqk   = (const float*)d_in[4]  + (size_t)dL * CC * 2 * CC;
    const float* W_proj  = (const float*)d_in[5]  + (size_t)dL * CC * CC;
    const float* b_proj  = (const float*)d_in[6]  + (size_t)dL * CC;
    const float* g_a1    = (const float*)d_in[7]  + (size_t)dL * CC;
    const float* be_a1   = (const float*)d_in[8]  + (size_t)dL * CC;
    const float* g_a2    = (const float*)d_in[9]  + (size_t)dL * CC;
    const float* be_a2   = (const float*)d_in[10] + (size_t)dL * CC;
    const float* g_j     = (const float*)d_in[11] + (size_t)dL * CC;
    const float* be_j    = (const float*)d_in[12] + (size_t)dL * CC;
    const float* g_r1    = (const float*)d_in[13] + (size_t)dL * 4 * CC;
    const float* be_r1   = (const float*)d_in[14] + (size_t)dL * 4 * CC;
    const float* g_r2    = (const float*)d_in[15] + (size_t)dL * CC;
    const float* be_r2   = (const float*)d_in[16] + (size_t)dL * CC;
    const float* Wj1     = (const float*)d_in[17] + (size_t)dL * CC * CC;
    const float* bj1     = (const float*)d_in[18] + (size_t)dL * CC;
    const float* Wj2     = (const float*)d_in[19] + (size_t)dL * CC * CC;
    const float* bj2     = (const float*)d_in[20] + (size_t)dL * CC;
    const float* Wr1a    = (const float*)d_in[21] + (size_t)dL * 16 * CC * CC;
    const float* br1a    = (const float*)d_in[22] + (size_t)dL * 4 * CC;
    const float* Wr1b    = (const float*)d_in[23] + (size_t)dL * 4 * CC * CC;
    const float* br1b    = (const float*)d_in[24] + (size_t)dL * CC;
    const float* Wr2a    = (const float*)d_in[25] + (size_t)dL * CC * CC;
    const float* br2a    = (const float*)d_in[26] + (size_t)dL * CC;
    const float* Wr2b    = (const float*)d_in[27] + (size_t)dL * CC * CC;
    const float* br2b    = (const float*)d_in[28] + (size_t)dL * CC;

    // repack this layer's weights into WMMA fragment layout
    PACK(W_jqkv, pw_jqkv, 128, 384);
    PACK(W_rqk,  pw_rqk,  128, 256);
    PACK(W_rconv,pw_rconv,128, 8);
    PACK(W_proj, pw_proj, 128, 128);
    PACK(Wj1,    pw_j1,   128, 128);
    PACK(Wj2,    pw_j2,   128, 128);
    PACK(Wr1a,   pw_r1a,  512, 512);
    PACK(Wr1b,   pw_r1b,  512, 128);
    PACK(Wr2a,   pw_r2a,  128, 128);
    PACK(Wr2b,   pw_r2b,  128, 128);

    // ---- attention ----
    ln_rows<128><<<64, 256, 0, stream>>>(jf, g_a1, be_a1, x_ln, BB * NN);
    GEMM(0, x_ln, pw_jqkv, nullptr, nullptr, jqkv, nullptr, BB * NN, 384, 128);
    ln_rows<128><<<RROWS / 8, 256, 0, stream>>>(rf, g_a2, be_a2, rA, RROWS);
    GEMM(1, rA, pw_rqk,   nullptr, nullptr, nullptr, rB, RROWS, 256, 128);
    GEMM(0, rA, pw_rconv, nullptr, nullptr, rlin, nullptr, RROWS, 8, 128);
    attn_softmax<<<BB * HH * NN, NN, 0, stream>>>(jqkv, rlin, rB, attn);
    attn_out<<<BB * NN * CC / 256, 256, 0, stream>>>(attn, jqkv, o_bf);
    GEMM(3, o_bf, pw_proj, b_proj, jf, jf, nullptr, BB * NN, 128, 128);

    // ---- joint MLP ----
    ln_rows<128><<<64, 256, 0, stream>>>(jf, g_j, be_j, y_bf, BB * NN);
    GEMM(2, y_bf, pw_j1, bj1, nullptr, nullptr, h_bf, BB * NN, 128, 128);
    GEMM(3, h_bf, pw_j2, bj2, jf, jf, nullptr, BB * NN, 128, 128);

    // ---- relation MLP 1 (the big GEMM) ----
    build_rin_ln<<<RROWS / 8, 256, 0, stream>>>(rf, jf, g_r1, be_r1, rA);
    GEMM(2, rA, pw_r1a, br1a, nullptr, nullptr, rB, RROWS, 512, 512);
    GEMM(3, rB, pw_r1b, br1b, rf, rf, nullptr, RROWS, 128, 512);

    // ---- relation MLP 2 ----
    ln_rows<128><<<RROWS / 8, 256, 0, stream>>>(rf, g_r2, be_r2, rA, RROWS);
    GEMM(2, rA, pw_r2a, br2a, nullptr, nullptr, rB, RROWS, 128, 128);
    GEMM(3, rB, pw_r2b, br2b, rf, rf, nullptr, RROWS, 128, 128);
  }
}
